// STU_42305427865774
// MI455X (gfx1250) — matrix operand order repaired and checked by reading, not compiled, not run on hardware
//
#include <hip/hip_runtime.h>
#include <stdint.h>

#define SEQ   2048
#define DIN   512
#define DOUT  512
#define KF    24
#define KU    3
#define KKTOT 48                    // 24 filters x {plus, minus}
#define KCOLS (KKTOT * DIN)         // 24576 spectral contraction cols
#define KTOT  (KCOLS + KU * DIN)    // 26112 total contraction dim (spectral + AR)

// workspace byte offsets (16B aligned); total ~130 MB
#define OFF_PHI 0u                  // 48 x 2048 bf16
#define OFF_XT  196608u             // xT: 512 x 2048 bf16 (x transposed)
#define OFF_M   2293760u            // MallT: 512 x 26112 bf16 (weights transposed)
#define OFF_U   29032448u           // Uall: 2048 x 26112 bf16

#if __has_builtin(__builtin_amdgcn_global_load_async_to_lds_b128) && \
    __has_builtin(__builtin_amdgcn_s_wait_asynccnt)
#define HAVE_ASYNC 1
#else
#define HAVE_ASYNC 0
#endif

typedef __bf16 bf16_t;
typedef __attribute__((ext_vector_type(16))) bf16_t       v16bf;
typedef __attribute__((ext_vector_type(8)))  float        v8f;
typedef __attribute__((ext_vector_type(4)))  unsigned int u32x4;

union Frag16 { v16bf v; unsigned short u[16]; u32x4 q[2]; };

__device__ __forceinline__ unsigned short f2bf(float f) {
    unsigned u = __float_as_uint(f);
    return (unsigned short)((u + 0x7FFFu + ((u >> 16) & 1u)) >> 16);  // RNE
}

#if HAVE_ASYNC
// builtin signature (verified via hipcc diagnostic): (v4i as1*, v4i as3*, Ii, Ii)
typedef __attribute__((__vector_size__(16))) int v4i_t;
typedef __attribute__((address_space(1))) v4i_t* gp128_t;
typedef __attribute__((address_space(3))) v4i_t* lp128_t;
#define ASYNC_WAIT(n) __builtin_amdgcn_s_wait_asynccnt(n)
#else
#define ASYNC_WAIT(n) ((void)0)
#endif

// Stage a 64-row x 32-bf16 tile (row pitch = rowStride elems) into LDS laid out
// [row][32 + 8 pad]. 128-thread WG, 2 x b128 per thread; async DMA if available.
__device__ __forceinline__ void stage64x32(unsigned short* lds,
                                           const unsigned short* g,
                                           long long rowStride, int tid) {
    #pragma unroll
    for (int i = 0; i < 2; ++i) {
        int e = tid + 128 * i;              // 256 x 16B = 4 KB
        int m = e >> 2, c = e & 3;
        const unsigned short* src = g + (long long)m * rowStride + c * 8;
        unsigned short*       dst = lds + m * 40 + c * 8;
#if HAVE_ASYNC
        __builtin_amdgcn_global_load_async_to_lds_b128(
            (gp128_t)(unsigned long long)src,
            (lp128_t)(unsigned int)(unsigned long long)dst, 0, 0);
#else
        *reinterpret_cast<u32x4*>(dst) = *reinterpret_cast<const u32x4*>(src);
#endif
    }
}

// ---------------------------------------------------------------------------
// Kernel P: bf16-pack phi (+/- sign-modulated), x transposed, fused weights
// transposed, and the AR (shifted-x) columns of U_all.
// ---------------------------------------------------------------------------
__global__ void stu_prep(const float* __restrict__ x, const float* __restrict__ phi,
                         const float* __restrict__ M, const float* __restrict__ Mp,
                         const float* __restrict__ Mm,
                         unsigned short* __restrict__ ws_phi,
                         unsigned short* __restrict__ xT,
                         unsigned short* __restrict__ MallT,
                         unsigned short* __restrict__ Uall) {
    long long idx = (long long)blockIdx.x * blockDim.x + threadIdx.x;
    const long long N_PHI = (long long)KKTOT * SEQ;
    const long long N_XT  = (long long)DIN * SEQ;
    const long long N_M   = (long long)DOUT * KTOT;
    const long long N_AR  = (long long)SEQ * (KU * DIN);

    if (idx < N_PHI) {
        int kk = (int)(idx / SEQ), s = (int)(idx % SEQ);
        float v = phi[s * KF + (kk % KF)];
        if (kk >= KF && (s & 1)) v = -v;          // phi_minus[s] = (-1)^s phi[s]
        ws_phi[idx] = f2bf(v);
        return;
    }
    idx -= N_PHI;
    if (idx < N_XT) {                             // xT[d][s] = x[s][d]
        int d = (int)(idx / SEQ), s = (int)(idx % SEQ);
        xT[idx] = f2bf(x[(long long)s * DIN + d]);
        return;
    }
    idx -= N_XT;
    if (idx < N_M) {                              // MallT[o][r]
        int o = (int)(idx / KTOT), r = (int)(idx % KTOT);
        float v;
        if (r < KF * DIN)            v = Mp[(long long)r * DOUT + o];
        else if (r < 2 * KF * DIN)   v = Mm[(long long)(r - KF * DIN) * DOUT + o];
        else {
            int rr = r - 2 * KF * DIN;
            int u = rr / DIN, i = rr % DIN;
            v = M[((long long)o * DIN + i) * KU + u];   // M is (d_out, d_in, k_u)
        }
        MallT[idx] = f2bf(v);
        return;
    }
    idx -= N_M;
    if (idx < N_AR) {                             // AR columns of U_all
        int t = (int)(idx / (KU * DIN));
        int rem = (int)(idx % (KU * DIN));
        int u = rem / DIN, d = rem % DIN;
        float v = (t >= u) ? x[(long long)(t - u) * DIN + d] : 0.0f;
        Uall[(long long)t * KTOT + KCOLS + u * DIN + d] = f2bf(v);
    }
}

// ---------------------------------------------------------------------------
// Kernel T: causal Toeplitz convolution as WMMA GEMM, PAIRED plus/minus:
//   U[t, (kk+p*24)*512+d] = sum_{s<=t} phi_{kk,p}[t-s] * x[s, d],  p in {0,1}
// Both sign branches share the x tiles, halving global->LDS traffic.
// phi columns stored REVERSED in LDS (zero tail = causal mask) -> A-fragments
// are ascending contiguous runs (pure ds_load_b128). x tiles triple-buffered
// via async DMA with a single barrier per K-step.
// ---------------------------------------------------------------------------
__global__ void __launch_bounds__(128)
stu_conv(const unsigned short* __restrict__ ws_phi,
         const unsigned short* __restrict__ xT,
         unsigned short* __restrict__ Uall) {
    __shared__ unsigned short ldsPhiR[2][SEQ + 64];  // [phi reversed | 64 zeros]
    __shared__ unsigned short ldsX[3][64 * 40];      // xT tile [d][s], triple buffer

    const int tid  = threadIdx.x;
    const int lane = tid & 31, w = tid >> 5;
    const int half = lane >> 4, ln = lane & 15;
    const int d0 = blockIdx.x * 64, t0 = blockIdx.y * 64, kk = blockIdx.z; // kk<24
    const int wm = w >> 1, wn = w & 1;
    const int tBase = t0 + wm * 32, dBase = d0 + wn * 32;

    #pragma unroll
    for (int p = 0; p < 2; ++p)
        for (int i = tid; i < SEQ + 64; i += 128)
            ldsPhiR[p][i] = (i < SEQ) ? ws_phi[(kk + p * KF) * SEQ + (SEQ - 1 - i)]
                                      : (unsigned short)0;

    const int s_end = (t0 + 64 < SEQ) ? (t0 + 64) : SEQ;   // causality: s <= max t
    stage64x32(ldsX[0], xT + (long long)d0 * SEQ, SEQ, tid);

    v8f acc[2][2][2] = {};                        // [p][fm][fn]
    int cur = 0, nxt = 1;
    for (int s0 = 0; s0 < s_end; s0 += 32) {
        if (s0 + 32 < s_end) {
            stage64x32(ldsX[nxt], xT + (long long)d0 * SEQ + (s0 + 32), SEQ, tid);
            ASYNC_WAIT(2);   // current tile landed; next tile stays in flight
        } else {
            ASYNC_WAIT(0);
        }
        __syncthreads();     // single barrier: buffer written at step j was last
                             // read at step j-2, separated by step j-1's barrier

        // A[t, s0+K] = phiR[2047 - t + s0 + K]; ISA A layout K(e,half):
        // e<8 -> 8*half + e ; e>=8 -> 8*half + e + 8  (ascending runs of 8)
        Frag16 a[2][2], b[2];
        #pragma unroll
        for (int fm = 0; fm < 2; ++fm) {
            int B0 = (SEQ - 1) - (tBase + fm * 16 + ln) + s0 + 8 * half;
            #pragma unroll
            for (int p = 0; p < 2; ++p)
                #pragma unroll
                for (int e = 0; e < 16; ++e) {
                    int off = (e < 8) ? e : (e + 8);
                    a[p][fm].u[e] = ldsPhiR[p][B0 + off];
                }
        }
        #pragma unroll
        for (int fn = 0; fn < 2; ++fn) {
            int dl = wn * 32 + fn * 16 + ln;
            const u32x4* pp = reinterpret_cast<const u32x4*>(&ldsX[cur][dl * 40 + half * 8]);
            b[fn].q[0] = pp[0];
            b[fn].q[1] = pp[2];
        }
        #pragma unroll
        for (int p = 0; p < 2; ++p)
            #pragma unroll
            for (int fm = 0; fm < 2; ++fm)
                #pragma unroll
                for (int fn = 0; fn < 2; ++fn)
                    acc[p][fm][fn] = __builtin_amdgcn_wmma_f32_16x16x32_bf16(
                        false, a[p][fm].v, false, b[fn].v, (short)0,
                        acc[p][fm][fn], false, false);

        cur = nxt; nxt = (nxt == 2) ? 0 : nxt + 1;
    }

    // C/D layout: vgpr r, lanes 0-15 -> M=r, lanes 16-31 -> M=r+8, N=lane&15
    #pragma unroll
    for (int p = 0; p < 2; ++p)
        #pragma unroll
        for (int fm = 0; fm < 2; ++fm)
            #pragma unroll
            for (int fn = 0; fn < 2; ++fn)
                #pragma unroll
                for (int r = 0; r < 8; ++r) {
                    int t = tBase + fm * 16 + r + 8 * half;
                    int d = dBase + fn * 16 + ln;
                    Uall[(long long)t * KTOT + (kk + p * KF) * DIN + d] =
                        f2bf(acc[p][fm][fn][r]);
                }
}

// ---------------------------------------------------------------------------
// Kernel G: out (2048x512 f32) = U_all (2048x26112) @ M_all (26112x512), bf16.
// Both operand tiles triple-buffered in LDS via async DMA (B pre-transposed).
// ---------------------------------------------------------------------------
__global__ void __launch_bounds__(128)
stu_gemm(const unsigned short* __restrict__ Uall,
         const unsigned short* __restrict__ MallT,
         float* __restrict__ out) {
    __shared__ unsigned short ldsA[3][64 * 40];   // A tile [m][k]
    __shared__ unsigned short ldsB[3][64 * 40];   // B tile [n][k]

    const int tid  = threadIdx.x;
    const int lane = tid & 31, w = tid >> 5;
    const int half = lane >> 4, ln = lane & 15;
    const int n0 = blockIdx.x * 64, t0 = blockIdx.y * 64;
    const int wm = w >> 1, wn = w & 1;

    stage64x32(ldsA[0], Uall  + (long long)t0 * KTOT, KTOT, tid);
    stage64x32(ldsB[0], MallT + (long long)n0 * KTOT, KTOT, tid);

    v8f acc[2][2] = {};
    int cur = 0, nxt = 1;
    for (int k0 = 0; k0 < KTOT; k0 += 32) {
        if (k0 + 32 < KTOT) {
            stage64x32(ldsA[nxt], Uall  + (long long)t0 * KTOT + (k0 + 32), KTOT, tid);
            stage64x32(ldsB[nxt], MallT + (long long)n0 * KTOT + (k0 + 32), KTOT, tid);
            ASYNC_WAIT(4);
#if !HAVE_ASYNC
            __builtin_prefetch(&Uall[(long long)(t0 + (tid >> 2)) * KTOT + k0 + 64], 0, 1);
            __builtin_prefetch(&MallT[(long long)(n0 + (tid >> 2)) * KTOT + k0 + 64], 0, 1);
#endif
        } else {
            ASYNC_WAIT(0);
        }
        __syncthreads();

        Frag16 a[2], b[2];
        #pragma unroll
        for (int fm = 0; fm < 2; ++fm) {
            int m = wm * 32 + fm * 16 + ln;
            a[fm].q[0] = *reinterpret_cast<const u32x4*>(&ldsA[cur][m * 40 + 8 * half]);
            a[fm].q[1] = *reinterpret_cast<const u32x4*>(&ldsA[cur][m * 40 + 16 + 8 * half]);
        }
        #pragma unroll
        for (int fn = 0; fn < 2; ++fn) {
            int n = wn * 32 + fn * 16 + ln;
            const u32x4* pb = reinterpret_cast<const u32x4*>(&ldsB[cur][n * 40 + half * 8]);
            b[fn].q[0] = pb[0];
            b[fn].q[1] = pb[2];
        }
        #pragma unroll
        for (int fm = 0; fm < 2; ++fm)
            #pragma unroll
            for (int fn = 0; fn < 2; ++fn)
                acc[fm][fn] = __builtin_amdgcn_wmma_f32_16x16x32_bf16(
                    false, a[fm].v, false, b[fn].v, (short)0, acc[fm][fn], false, false);

        cur = nxt; nxt = (nxt == 2) ? 0 : nxt + 1;
    }

    #pragma unroll
    for (int fm = 0; fm < 2; ++fm)
        #pragma unroll
        for (int fn = 0; fn < 2; ++fn)
            #pragma unroll
            for (int r = 0; r < 8; ++r) {
                int t = t0 + wm * 32 + fm * 16 + r + 8 * half;
                int o = n0 + wn * 32 + fn * 16 + ln;
                out[(long long)t * DOUT + o] = acc[fm][fn][r];
            }
}

// ---------------------------------------------------------------------------
extern "C" void kernel_launch(void* const* d_in, const int* in_sizes, int n_in,
                              void* d_out, int out_size, void* d_ws, size_t ws_size,
                              hipStream_t stream) {
    (void)in_sizes; (void)n_in; (void)out_size; (void)ws_size;
    const float* x   = (const float*)d_in[0];
    const float* phi = (const float*)d_in[1];
    const float* M   = (const float*)d_in[2];
    const float* Mp  = (const float*)d_in[3];
    const float* Mm  = (const float*)d_in[4];

    char* ws = (char*)d_ws;
    unsigned short* ws_phi = (unsigned short*)(ws + OFF_PHI);
    unsigned short* xT     = (unsigned short*)(ws + OFF_XT);
    unsigned short* MallT  = (unsigned short*)(ws + OFF_M);
    unsigned short* Uall   = (unsigned short*)(ws + OFF_U);

    long long totalPrep = (long long)KKTOT * SEQ + (long long)DIN * SEQ +
                          (long long)DOUT * KTOT + (long long)SEQ * (KU * DIN);
    int pblocks = (int)((totalPrep + 255) / 256);

    stu_prep<<<pblocks, 256, 0, stream>>>(x, phi, M, Mp, Mm, ws_phi, xT, MallT, Uall);
    stu_conv<<<dim3(8, 32, KF), 128, 0, stream>>>(ws_phi, xT, Uall);
    stu_gemm<<<dim3(8, 32), 128, 0, stream>>>(Uall, MallT, (float*)d_out);
}
